// CustomSwin_15994458211318
// MI455X (gfx1250) — compile-verified, matrix operand order checked by reading, not run on hardware
//
#include <hip/hip_runtime.h>
#include <hip/hip_bf16.h>
#include <math.h>

// ---------------------------------------------------------------------------
// CDNA5 (gfx1250) Swin-T forward.
//  - all GEMM work via v_wmma_f32_16x16x32_f16 (f16 in, f32 accumulate)
//  - GEMM tiles staged global->LDS with global_load_async_to_lds_b128,
//    double-buffered (ASYNCcnt pipelining)
//  - f16 activation/weight arenas, f32 residual stream
// ---------------------------------------------------------------------------

typedef _Float16 v16h __attribute__((ext_vector_type(16)));
typedef float    v8f  __attribute__((ext_vector_type(8)));

#define WMMA16(a, b, c) \
  __builtin_amdgcn_wmma_f32_16x16x32_f16(false, (a), false, (b), (short)0, (c), false, false)

// ======================= generic WMMA GEMM =================================
// C[M,N] = act(A[M,K] @ W[N,K]^T + bias) (+ resid). A,W are f16 row-major.
// Host guarantees: M % 64 == 0, K % 32 == 0, 16B-aligned pointers.
// 256 threads = 8 waves; 64x128 block tile; each wave: 16x64 strip = 4 WMMA
// accumulators, A-fragment reused 4x per K-step. LDS double-buffered with
// async global->LDS copies kept 3-deep in flight.
template <typename OutT>
__global__ __launch_bounds__(256) void gemm_wmma_kernel(
    const _Float16* __restrict__ A, const _Float16* __restrict__ Wt,
    const float* __restrict__ bias, const float* __restrict__ resid,
    OutT* __restrict__ Cout, int M, int N, int K, int do_gelu) {
  // row stride 40 halfs = 80B: every row start is 16B aligned for b128 async
  __shared__ _Float16 As[2][64][40];
  __shared__ _Float16 Bs[2][128][40];

  const int tid  = threadIdx.x;
  const int m0   = blockIdx.x * 64;
  const int n0   = blockIdx.y * 128;
  const int wave = tid >> 5;
  const int lane = tid & 31;
  const int l16  = lane & 15;
  const int hi   = lane >> 4;  // 16-lane half
  const int kofs = hi * 8;     // ISA 16-bit A/B fragment K sub-offset
  const int msub = (wave >> 1) * 16;
  const int nsub = (wave & 1) * 64;

  // per-thread load slots: A = one b128 (64x32 tile), B = two b128 (128x32)
  const int lrA = tid >> 2;          // 0..63
  const int lcA = (tid & 3) * 8;     // 0,8,16,24
  const int lrB = tid >> 1;          // 0..127
  const int lcB = (tid & 1) * 16;    // 0,16
  const int gnB = (n0 + lrB < N) ? (n0 + lrB) : (N - 1);  // clamp; cols guarded
  const _Float16* aRow = A + (size_t)(m0 + lrA) * K + lcA;
  const _Float16* bRow = Wt + (size_t)gnB * K + lcB;
  const unsigned ldsA0 = (unsigned)(uintptr_t)&As[0][lrA][lcA];
  const unsigned ldsA1 = (unsigned)(uintptr_t)&As[1][lrA][lcA];
  const unsigned ldsB0 = (unsigned)(uintptr_t)&Bs[0][lrB][lcB];
  const unsigned ldsB1 = (unsigned)(uintptr_t)&Bs[1][lrB][lcB];

#define GEMM_ISSUE(bufsel, kk)                                              \
  do {                                                                      \
    asm volatile("global_load_async_to_lds_b128 %0, %1, off"                \
                 :: "v"((bufsel) ? ldsA1 : ldsA0), "v"(aRow + (kk))         \
                 : "memory");                                               \
    asm volatile("global_load_async_to_lds_b128 %0, %1, off"                \
                 :: "v"((bufsel) ? ldsB1 : ldsB0), "v"(bRow + (kk))         \
                 : "memory");                                               \
    asm volatile("global_load_async_to_lds_b128 %0, %1, off"                \
                 :: "v"(((bufsel) ? ldsB1 : ldsB0) + 16u),                  \
                    "v"(bRow + (kk) + 8)                                    \
                 : "memory");                                               \
  } while (0)

  v8f acc[4] = {v8f{}, v8f{}, v8f{}, v8f{}};

  const int nT = K >> 5;
  GEMM_ISSUE(0, 0);
  for (int t = 0; t < nT; ++t) {
    const int cb = t & 1;
    if (t + 1 < nT) {
      GEMM_ISSUE((t + 1) & 1, (t + 1) << 5);
      asm volatile("s_wait_asynccnt 0x3" ::: "memory");  // current tile landed
    } else {
      asm volatile("s_wait_asynccnt 0x0" ::: "memory");
    }
    __syncthreads();

    v16h a;
    const int ar = msub + l16;
#pragma unroll
    for (int j = 0; j < 8; ++j) {
      a[j]     = As[cb][ar][kofs + j];
      a[8 + j] = As[cb][ar][16 + kofs + j];
    }
#pragma unroll
    for (int nt = 0; nt < 4; ++nt) {
      v16h b;
      const int br = nsub + nt * 16 + l16;
#pragma unroll
      for (int j = 0; j < 8; ++j) {
        b[j]     = Bs[cb][br][kofs + j];
        b[8 + j] = Bs[cb][br][16 + kofs + j];
      }
      acc[nt] = WMMA16(a, b, acc[nt]);
    }
    __syncthreads();
  }
#undef GEMM_ISSUE

  // epilogue: D layout = 8 VGPRs, row m = msub + r + hi*8, col n = base + l16
#pragma unroll
  for (int r = 0; r < 8; ++r) {
    const int m = m0 + msub + r + hi * 8;
#pragma unroll
    for (int nt = 0; nt < 4; ++nt) {
      const int n = n0 + nsub + nt * 16 + l16;
      if (n >= N) continue;
      float v = acc[nt][r];
      if (bias) v += bias[n];
      if (do_gelu) v = 0.5f * v * (1.0f + erff(v * 0.70710678118654752f));
      if (resid) v += resid[(size_t)m * N + n];
      Cout[(size_t)m * N + n] = (OutT)v;
    }
  }
}

// ======================= fused window attention ============================
// One wave per (window, head). N = 49 padded to 64, head dim d = 32.
// Roll / window-partition / win_rev / un-roll baked into addressing.
// Per-token lookups (rel-pos index, shift-mask region, scatter coords) are
// precomputed once into LDS tables; the hot epilogue is LDS read + add.
__device__ __forceinline__ int swin_region(int g, int L) {
  return (g < L - 7) ? 0 : ((g < L - 3) ? 1 : 2);
}

__global__ __launch_bounds__(32) void win_attn_kernel(
    const _Float16* __restrict__ qkv, _Float16* __restrict__ att,
    const float* __restrict__ rpb, int B, int H, int W, int C, int nh,
    int shift) {
  const int nWw  = W / 7;
  const int nW   = (H / 7) * nWw;
  const int win  = blockIdx.x;  // b*nW + wi
  const int head = blockIdx.y;
  const int b    = win / nW;
  const int wi   = win % nW;
  const int wh   = wi / nWw;
  const int ww   = wi % nWw;
  const int lane = threadIdx.x;
  const int C3   = 3 * C;
  const int l16  = lane & 15;
  const int hi   = lane >> 4;
  const int kofs = hi * 8;

  __shared__ _Float16 Qs[64][40];
  __shared__ _Float16 Ks[64][40];
  __shared__ _Float16 Vs[64][40];
  __shared__ float    Sc[64][65];
  __shared__ _Float16 Ps[64][72];
  __shared__ float    Rp[169];  // per-head relative position bias table
  __shared__ int      Ta[49];   // r*13 + c   (rel = Ta[i]-Ta[j]+84)
  __shared__ int      Tid[49];  // shift-mask region id
  __shared__ int      Tsh[49];  // pre-rolled source/dest row
  __shared__ int      Tsw[49];  // pre-rolled source/dest col

  // preload this head's 169 rpb values + per-token tables
  for (int e = lane; e < 169; e += 32) Rp[e] = rpb[e * nh + head];
  for (int e = lane; e < 49; e += 32) {
    const int r = e / 7, c = e % 7;
    Ta[e] = r * 13 + c;
    Tid[e] = swin_region(wh * 7 + r, H) * 3 + swin_region(ww * 7 + c, W);
    int sh = wh * 7 + r + shift; if (sh >= H) sh -= H;
    int sw = ww * 7 + c + shift; if (sw >= W) sw -= W;
    Tsh[e] = sh; Tsw[e] = sw;
  }
  __syncthreads();

  // gather Q/K/V for this (window, head); roll baked into source addr
  for (int e = lane; e < 64 * 32; e += 32) {
    const int row = e >> 5, d = e & 31;
    _Float16 qh = (_Float16)0.f, kh = (_Float16)0.f, vh = (_Float16)0.f;
    if (row < 49) {
      const size_t t = ((size_t)(b * H + Tsh[row])) * W + Tsw[row];
      const _Float16* p = qkv + t * C3 + head * 32 + d;
      qh = (_Float16)((float)p[0] * 0.17677669529663687f);  // 1/sqrt(32)
      kh = p[C];
      vh = p[2 * C];
    }
    Qs[row][d] = qh; Ks[row][d] = kh; Vs[row][d] = vh;
  }
  __syncthreads();

  // scores = Q K^T (+rpb, +mask): 4x4 tiles of 16x16, K = 32
  for (int mi = 0; mi < 4; ++mi) {
    v16h a;
    const int mrow = mi * 16 + l16;
#pragma unroll
    for (int j = 0; j < 8; ++j) {
      a[j]     = Qs[mrow][kofs + j];
      a[8 + j] = Qs[mrow][16 + kofs + j];
    }
    for (int ni = 0; ni < 4; ++ni) {
      v16h bf;
      const int nrow = ni * 16 + l16;
#pragma unroll
      for (int j = 0; j < 8; ++j) {
        bf[j]     = Ks[nrow][kofs + j];
        bf[8 + j] = Ks[nrow][16 + kofs + j];
      }
      v8f acc = {};
      acc = WMMA16(a, bf, acc);
#pragma unroll
      for (int r = 0; r < 8; ++r) {
        const int i  = mi * 16 + r + hi * 8;
        const int j2 = ni * 16 + l16;
        float s = -1e30f;
        if (i < 49 && j2 < 49) {
          s = acc[r] + Rp[Ta[i] - Ta[j2] + 84];
          if (shift && (Tid[i] != Tid[j2])) s -= 100.0f;
        }
        Sc[i][j2] = s;
      }
    }
  }
  __syncthreads();

  // softmax over 49 valid columns; write normalized probs as f16
  for (int rsel = 0; rsel < 2; ++rsel) {
    const int row = lane + rsel * 32;
    const bool valid = row < 49;
    float mx = -1e30f, sum = 0.0f;
    if (valid) {
      for (int j = 0; j < 49; ++j) mx = fmaxf(mx, Sc[row][j]);
      for (int j = 0; j < 49; ++j) sum += __expf(Sc[row][j] - mx);
    }
    const float inv = (valid && sum > 0.0f) ? 1.0f / sum : 0.0f;
    for (int j = 0; j < 64; ++j) {
      float p = 0.0f;
      if (valid && j < 49) p = __expf(Sc[row][j] - mx) * inv;
      Ps[row][j] = (_Float16)p;
    }
  }
  __syncthreads();

  // out = P V : M=64, N=32, K=64 (two K=32 steps); scatter with un-roll
  for (int mi = 0; mi < 4; ++mi) {
    v16h a0, a1;
    const int mrow = mi * 16 + l16;
#pragma unroll
    for (int j = 0; j < 8; ++j) {
      a0[j]     = Ps[mrow][kofs + j];
      a0[8 + j] = Ps[mrow][16 + kofs + j];
      a1[j]     = Ps[mrow][32 + kofs + j];
      a1[8 + j] = Ps[mrow][48 + kofs + j];
    }
    for (int ni = 0; ni < 2; ++ni) {
      v16h b0, b1;
      const int n = ni * 16 + l16;
#pragma unroll
      for (int j = 0; j < 8; ++j) {
        b0[j]     = Vs[kofs + j][n];
        b0[8 + j] = Vs[16 + kofs + j][n];
        b1[j]     = Vs[32 + kofs + j][n];
        b1[8 + j] = Vs[48 + kofs + j][n];
      }
      v8f acc = {};
      acc = WMMA16(a0, b0, acc);
      acc = WMMA16(a1, b1, acc);
#pragma unroll
      for (int r = 0; r < 8; ++r) {
        const int i = mi * 16 + r + hi * 8;
        if (i >= 49) continue;
        const int d = ni * 16 + l16;
        att[(((size_t)(b * H + Tsh[i])) * W + Tsw[i]) * C + head * 32 + d] =
            (_Float16)acc[r];
      }
    }
  }
}

// ======================= LayerNorm (one wave / token) ======================
template <typename Tin, typename Tout>
__global__ __launch_bounds__(256) void ln_kernel(
    const Tin* __restrict__ in, Tout* __restrict__ out,
    const float* __restrict__ g, const float* __restrict__ bta, long T, int C) {
  const int wave = threadIdx.x >> 5;
  const int lane = threadIdx.x & 31;
  const long t = (long)blockIdx.x * 8 + wave;
  if (t >= T) return;
  const Tin* x = in + (size_t)t * C;
  float s = 0.0f;
  for (int c = lane; c < C; c += 32) s += (float)x[c];
  for (int o = 16; o > 0; o >>= 1) s += __shfl_xor(s, o, 32);
  const float m = s / (float)C;
  float v = 0.0f;
  for (int c = lane; c < C; c += 32) { const float d = (float)x[c] - m; v += d * d; }
  for (int o = 16; o > 0; o >>= 1) v += __shfl_xor(v, o, 32);
  const float rs = rsqrtf(v / (float)C + 1e-5f);
  Tout* y = out + (size_t)t * C;
  for (int c = lane; c < C; c += 32)
    y[c] = (Tout)(((float)x[c] - m) * rs * g[c] + bta[c]);
}

// ======================= converters / gathers / pool =======================
__global__ void cvt_f32_f16_kernel(const float* __restrict__ in,
                                   _Float16* __restrict__ out, long n) {
  const long i = (long)blockIdx.x * blockDim.x + threadIdx.x;
  if (i < n) out[i] = (_Float16)in[i];
}

// pe_w [96,48] -> f16 [96,64] zero-padded (K padded to WMMA granularity)
__global__ void cvt_pad_pe_kernel(const float* __restrict__ in,
                                  _Float16* __restrict__ out) {
  const int i = blockIdx.x * blockDim.x + threadIdx.x;
  if (i >= 96 * 64) return;
  const int r = i >> 6, c = i & 63;
  out[i] = (c < 48) ? (_Float16)in[r * 48 + c] : (_Float16)0.f;
}

// patch embed gather: [B,3,224,224] f32 -> [B*3136, 64] f16 (48 real + 16 pad)
__global__ void patch_gather_kernel(const float* __restrict__ x,
                                    _Float16* __restrict__ out, long n) {
  const long i = (long)blockIdx.x * blockDim.x + threadIdx.x;
  if (i >= n) return;
  const int ch = (int)(i & 63);
  const long t = i >> 6;
  float v = 0.0f;
  if (ch < 48) {
    const int w = (int)(t % 56);
    const int h = (int)((t / 56) % 56);
    const int b = (int)(t / 3136);
    const int c = ch >> 4, dh = (ch >> 2) & 3, dw = ch & 3;
    v = x[(((size_t)(b * 3 + c) * 224) + (4 * h + dh)) * 224 + (4 * w + dw)];
  }
  out[i] = (_Float16)v;
}

// patch merge gather: [B,H,W,C] f32 -> [B,H/2,W/2,4C] f16,
// group order (a,b) = (0,0),(1,0),(0,1),(1,1)
__global__ void merge_gather_kernel(const float* __restrict__ in,
                                    _Float16* __restrict__ out, int B, int H,
                                    int W, int C, long n) {
  const long i = (long)blockIdx.x * blockDim.x + threadIdx.x;
  if (i >= n) return;
  const int C4 = 4 * C;
  const int ch4 = (int)(i % C4);
  const int w2  = (int)((i / C4) % (W / 2));
  const int h2  = (int)((i / ((long)C4 * (W / 2))) % (H / 2));
  const int b   = (int)(i / ((long)C4 * (W / 2) * (H / 2)));
  const int g = ch4 / C, c = ch4 % C;
  const int a = g & 1, bb = g >> 1;
  out[i] = (_Float16)
      in[(((size_t)(b * H + 2 * h2 + a)) * W + (2 * w2 + bb)) * C + c];
}

// mean over 49 tokens: [64,49,768] f32 -> [64,768] f32
__global__ void pool_kernel(const float* __restrict__ in,
                            float* __restrict__ out, long n) {
  const long i = (long)blockIdx.x * blockDim.x + threadIdx.x;
  if (i >= n) return;
  const int c = (int)(i % 768);
  const int b = (int)(i / 768);
  float s = 0.0f;
  for (int t = 0; t < 49; ++t) s += in[((size_t)(b * 49 + t)) * 768 + c];
  out[i] = s * (1.0f / 49.0f);
}

// ======================= host orchestration ================================
static inline void launch_gemm_f(const _Float16* A, const _Float16* Wt,
                                 const float* bias, const float* resid,
                                 float* C, int M, int N, int K, int gelu,
                                 hipStream_t st) {
  dim3 grid(M / 64, (N + 127) / 128);
  gemm_wmma_kernel<float><<<grid, 256, 0, st>>>(A, Wt, bias, resid, C, M, N, K, gelu);
}
static inline void launch_gemm_h(const _Float16* A, const _Float16* Wt,
                                 const float* bias, _Float16* C, int M, int N,
                                 int K, int gelu, hipStream_t st) {
  dim3 grid(M / 64, (N + 127) / 128);
  gemm_wmma_kernel<_Float16><<<grid, 256, 0, st>>>(A, Wt, bias, nullptr, C, M, N, K, gelu);
}
static inline void launch_cvt(const float* in, _Float16* out, long n,
                              hipStream_t st) {
  cvt_f32_f16_kernel<<<(int)((n + 255) / 256), 256, 0, st>>>(in, out, n);
}

// logical per-block param slots
enum { BN1W = 0, BN1B, BQKVW, BQKVB, BRPB, BPROJW, BPROJB, BN2W, BN2B,
       BFC1W, BFC1B, BFC2W, BFC2B };

extern "C" void kernel_launch(void* const* d_in, const int* in_sizes, int n_in,
                              void* d_out, int out_size, void* d_ws,
                              size_t ws_size, hipStream_t stream) {
  (void)n_in; (void)out_size; (void)ws_size;
  // -------- input index mapping (two possible pytree flattening orders) ----
  int blk[12][13], mrg[3][3];
  int I_pe_w, I_pe_b, I_pe_n_w, I_pe_n_b, I_norm_w, I_norm_b;
  int I_hn_w, I_hn_b, I_h1_w, I_h1_b, I_h2_w, I_h2_b, I_x;
  const bool insertion = (in_sizes[0] > 1000000);  // x first => insertion order
  if (insertion) {
    I_x = 0; I_pe_w = 1; I_pe_b = 2; I_pe_n_w = 3; I_pe_n_b = 4;
    for (int bi = 0; bi < 12; ++bi)
      for (int j = 0; j < 13; ++j) blk[bi][j] = 5 + bi * 13 + j;
    for (int m = 0; m < 3; ++m) {
      mrg[m][0] = 161 + m * 3;      // n_w
      mrg[m][1] = 161 + m * 3 + 1;  // n_b
      mrg[m][2] = 161 + m * 3 + 2;  // red_w
    }
    I_norm_w = 170; I_norm_b = 171; I_hn_w = 172; I_hn_b = 173;
    I_h1_w = 174; I_h1_b = 175; I_h2_w = 176; I_h2_b = 177;
  } else {  // jax tree_flatten: sorted dict keys, params before x
    for (int bi = 0; bi < 12; ++bi) {
      const int base = bi * 13;
      blk[bi][BFC1B] = base + 0;  blk[bi][BFC1W] = base + 1;
      blk[bi][BFC2B] = base + 2;  blk[bi][BFC2W] = base + 3;
      blk[bi][BN1B]  = base + 4;  blk[bi][BN1W]  = base + 5;
      blk[bi][BN2B]  = base + 6;  blk[bi][BN2W]  = base + 7;
      blk[bi][BPROJB]= base + 8;  blk[bi][BPROJW]= base + 9;
      blk[bi][BQKVB] = base + 10; blk[bi][BQKVW] = base + 11;
      blk[bi][BRPB]  = base + 12;
    }
    I_h1_b = 156; I_h1_w = 157; I_h2_b = 158; I_h2_w = 159;
    I_hn_b = 160; I_hn_w = 161;
    for (int m = 0; m < 3; ++m) {
      mrg[m][1] = 162 + m * 3;      // n_b
      mrg[m][0] = 162 + m * 3 + 1;  // n_w
      mrg[m][2] = 162 + m * 3 + 2;  // red_w
    }
    I_norm_b = 171; I_norm_w = 172; I_pe_b = 173; I_pe_n_b = 174;
    I_pe_n_w = 175; I_pe_w = 176; I_x = 177;
  }
  auto P = [&](int i) -> const float* { return (const float*)d_in[i]; };

  // -------- workspace layout ----------------------------------------------
  const long SZ_TOK = 19267584L;  // 200704 * 96 (max tokens*C)
  const long SZ_BIG = 77070336L;  // 200704 * 384 (max tokens*4C)
  char* base = (char*)d_ws;
  size_t off = 0;
  auto carve = [&](size_t bytes) {
    void* p = base + off;
    off = (off + bytes + 255) & ~(size_t)255;
    return p;
  };
  float*    bufH   = (float*)carve(SZ_TOK * 4);    // residual stream (f32)
  _Float16* h16A   = (_Float16*)carve(SZ_TOK * 2); // LN out / merge concat
  _Float16* h16B   = (_Float16*)carve(SZ_TOK * 2); // attn out / merge LN out
  _Float16* h16BIG = (_Float16*)carve(SZ_BIG * 2); // qkv / fc1 / patch gather
  _Float16* WhBase = (_Float16*)carve(34000000L * 2);  // f16 weight arena
  float*    bufNF  = (float*)carve(2408448L * 4);  // final LN out [64*49,768]
  float*    pool1  = (float*)carve(64 * 768 * 4);
  _Float16* pool2h = (_Float16*)carve(64 * 768 * 2);
  _Float16* headTh = (_Float16*)carve(64 * 256 * 2);

  // -------- f16 weight arena (converted every call; in-order stream) -------
  size_t cur = 0;
  auto halloc = [&](size_t n) { size_t o = cur; cur += (n + 15) & ~(size_t)15; return o; };
  size_t w_pe = halloc(96 * 64);
  cvt_pad_pe_kernel<<<(96 * 64 + 255) / 256, 256, 0, stream>>>(P(I_pe_w), WhBase + w_pe);

  const int B = 64;
  const int DIMS[4]   = {96, 192, 384, 768};
  const int HEADS[4]  = {3, 6, 12, 24};
  const int RES[4]    = {56, 28, 14, 7};
  const int DEPTHS[4] = {2, 2, 6, 2};

  size_t w_qkv[12], w_proj[12], w_fc1[12], w_fc2[12], w_red[3];
  {
    int bi = 0;
    for (int s = 0; s < 4; ++s) {
      const long C = DIMS[s];
      for (int d = 0; d < DEPTHS[s]; ++d) {
        w_qkv[bi]  = halloc(3 * C * C);
        w_proj[bi] = halloc(C * C);
        w_fc1[bi]  = halloc(4 * C * C);
        w_fc2[bi]  = halloc(4 * C * C);
        launch_cvt(P(blk[bi][BQKVW]), WhBase + w_qkv[bi], 3 * C * C, stream);
        launch_cvt(P(blk[bi][BPROJW]), WhBase + w_proj[bi], C * C, stream);
        launch_cvt(P(blk[bi][BFC1W]), WhBase + w_fc1[bi], 4 * C * C, stream);
        launch_cvt(P(blk[bi][BFC2W]), WhBase + w_fc2[bi], 4 * C * C, stream);
        ++bi;
      }
      if (s < 3) {
        w_red[s] = halloc(8 * C * C);
        launch_cvt(P(mrg[s][2]), WhBase + w_red[s], 8 * C * C, stream);
      }
    }
  }
  size_t w_h1 = halloc(256 * 768);
  size_t w_h2 = halloc(256);
  launch_cvt(P(I_h1_w), WhBase + w_h1, 256 * 768, stream);
  launch_cvt(P(I_h2_w), WhBase + w_h2, 256, stream);

  // -------- patch embed: gather(pad K=64) -> GEMM -> LN --------------------
  {
    const long n = (long)B * 3136 * 64;
    patch_gather_kernel<<<(int)((n + 255) / 256), 256, 0, stream>>>(
        P(I_x), h16BIG, n);
    launch_gemm_f(h16BIG, WhBase + w_pe, P(I_pe_b), nullptr, bufH,
                  B * 3136, 96, 64, 0, stream);
    ln_kernel<float, float><<<(B * 3136 + 7) / 8, 256, 0, stream>>>(
        bufH, bufH, P(I_pe_n_w), P(I_pe_n_b), (long)B * 3136, 96);
  }

  // -------- stages ---------------------------------------------------------
  int bi = 0;
  for (int s = 0; s < 4; ++s) {
    const int C = DIMS[s], nh = HEADS[s], H = RES[s];
    const long T = (long)B * H * H;
    for (int d = 0; d < DEPTHS[s]; ++d) {
      const int shift = (d % 2 == 1 && s < 3) ? 3 : 0;
      const int* q = blk[bi];
      // ---- attention branch
      ln_kernel<float, _Float16><<<(int)((T + 7) / 8), 256, 0, stream>>>(
          bufH, h16A, P(q[BN1W]), P(q[BN1B]), T, C);
      launch_gemm_h(h16A, WhBase + w_qkv[bi], P(q[BQKVB]), h16BIG,
                    (int)T, 3 * C, C, 0, stream);
      {
        dim3 grid(B * (H / 7) * (H / 7), nh);
        win_attn_kernel<<<grid, 32, 0, stream>>>(h16BIG, h16B, P(q[BRPB]),
                                                 B, H, H, C, nh, shift);
      }
      launch_gemm_f(h16B, WhBase + w_proj[bi], P(q[BPROJB]), bufH, bufH,
                    (int)T, C, C, 0, stream);
      // ---- mlp branch
      ln_kernel<float, _Float16><<<(int)((T + 7) / 8), 256, 0, stream>>>(
          bufH, h16A, P(q[BN2W]), P(q[BN2B]), T, C);
      launch_gemm_h(h16A, WhBase + w_fc1[bi], P(q[BFC1B]), h16BIG,
                    (int)T, 4 * C, C, 1, stream);
      launch_gemm_f(h16BIG, WhBase + w_fc2[bi], P(q[BFC2B]), bufH, bufH,
                    (int)T, C, 4 * C, 0, stream);
      ++bi;
    }
    if (s < 3) {  // ---- patch merge
      const long T2 = T / 4;
      const long n = T2 * 4 * C;
      merge_gather_kernel<<<(int)((n + 255) / 256), 256, 0, stream>>>(
          bufH, h16A, B, H, H, C, n);
      ln_kernel<_Float16, _Float16><<<(int)((T2 + 7) / 8), 256, 0, stream>>>(
          h16A, h16B, P(mrg[s][0]), P(mrg[s][1]), T2, 4 * C);
      launch_gemm_f(h16B, WhBase + w_red[s], nullptr, nullptr, bufH,
                    (int)T2, 2 * C, 4 * C, 0, stream);
    }
  }

  // -------- final norm, pool, head ----------------------------------------
  ln_kernel<float, float><<<(B * 49 + 7) / 8, 256, 0, stream>>>(
      bufH, bufNF, P(I_norm_w), P(I_norm_b), (long)B * 49, 768);
  {
    const long n = (long)B * 768;
    pool_kernel<<<(int)((n + 255) / 256), 256, 0, stream>>>(bufNF, pool1, n);
  }
  ln_kernel<float, _Float16><<<(B + 7) / 8, 256, 0, stream>>>(
      pool1, pool2h, P(I_hn_w), P(I_hn_b), B, 768);
  launch_gemm_h(pool2h, WhBase + w_h1, P(I_h1_b), headTh, B, 256, 768, 1,
                stream);
  launch_gemm_f(headTh, WhBase + w_h2, P(I_h2_b), nullptr, (float*)d_out,
                B, 1, 256, 0, stream);
}
